// FP8Linear_52347061404317
// MI455X (gfx1250) — 8-bit matrix operand order repaired and checked by reading, compiled, not run on hardware
//
#include <hip/hip_runtime.h>
#include <hip/hip_bf16.h>

#define MDIM 8192
#define KDIM 4096
#define NDIM 11008
#define FP8MAXF 448.0f

typedef __attribute__((ext_vector_type(16))) int   v16i;
typedef __attribute__((ext_vector_type(8)))  float v8f;

#if defined(__has_builtin)
#if __has_builtin(__builtin_amdgcn_cvt_pk_fp8_f32)
#define HAVE_CVT_PK_FP8 1
#endif
#endif


__global__ void init_scalars_kernel(unsigned int* s) {
    if (threadIdx.x < 2) s[threadIdx.x] = 0u;
}

__global__ __launch_bounds__(256) void amax_bf16_kernel(
    const uint4* __restrict__ in, long long n128, unsigned int* __restrict__ amax_bits) {
    float m = 0.0f;
    const long long stride = (long long)gridDim.x * blockDim.x;
    for (long long i = (long long)blockIdx.x * blockDim.x + threadIdx.x; i < n128; i += stride) {
        uint4 v = in[i];
        const unsigned int uu[4] = {v.x, v.y, v.z, v.w};
#pragma unroll
        for (int j = 0; j < 4; ++j) {
            m = fmaxf(m, __uint_as_float((uu[j] & 0x7FFFu) << 16));
            m = fmaxf(m, __uint_as_float(uu[j] & 0x7FFF0000u));
        }
    }
#pragma unroll
    for (int off = 16; off > 0; off >>= 1)
        m = fmaxf(m, __shfl_xor(m, off, 32));
    __shared__ float red[8];
    const int lane = threadIdx.x & 31;
    const int wv   = threadIdx.x >> 5;
    if (lane == 0) red[wv] = m;
    __syncthreads();
    if (threadIdx.x == 0) {
        float bm = red[0];
        for (int i = 1; i < (int)(blockDim.x >> 5); ++i) bm = fmaxf(bm, red[i]);
        atomicMax(amax_bits, __float_as_uint(bm));
    }
}

__global__ void compute_scales_kernel(unsigned int* sb) {
    const float ax = fmaxf(__uint_as_float(sb[0]), 1e-12f);
    const float aw = fmaxf(__uint_as_float(sb[1]), 1e-12f);
    float* sf = (float*)sb;
    sf[2] = FP8MAXF / ax;
    sf[3] = FP8MAXF / aw;
    sf[4] = (ax / FP8MAXF) * (aw / FP8MAXF);
}

#if !defined(HAVE_CVT_PK_FP8)
__device__ inline unsigned int sw_pack4_e4m3(float f0, float f1, float f2, float f3) {
    float f[4] = {f0, f1, f2, f3};
    unsigned int r = 0u;
#pragma unroll
    for (int i = 0; i < 4; ++i) {
        unsigned int u   = __float_as_uint(f[i]);
        unsigned int sgn = (u >> 24) & 0x80u;
        unsigned int a   = u & 0x7FFFFFFFu;
        unsigned int enc = 0u;
        if (a != 0u) {
            int exp = (int)(a >> 23) - 127;
            unsigned int man = (a & 0x7FFFFFu) | 0x800000u;
            int shift = (exp < -6) ? (20 + (-6 - exp)) : 20;
            if (shift <= 31) {
                unsigned int q   = man >> shift;
                unsigned int rem = man & ((1u << shift) - 1u);
                unsigned int h   = 1u << (shift - 1);
                if (rem > h || (rem == h && (q & 1u))) ++q;
                int eq; unsigned int mq;
                if (exp >= -6) {
                    eq = exp + 7;
                    if (q >= 16u) { q >>= 1; ++eq; }
                    mq = q & 7u;
                } else if (q >= 8u) { eq = 1; mq = 0u; }
                else                { eq = 0; mq = q; }
                if (eq > 15 || (eq == 15 && mq == 7u)) { eq = 15; mq = 6u; }
                enc = ((unsigned int)eq << 3) | mq;
            }
        }
        r |= (sgn | enc) << (8 * i);
    }
    return r;
}
#endif

__global__ __launch_bounds__(256) void quantize_bf16_fp8_kernel(
    const uint4* __restrict__ in, uint2* __restrict__ outq,
    long long n128, const float* __restrict__ sf, int slot) {
    const float s = sf[slot];
    const long long stride = (long long)gridDim.x * blockDim.x;
    for (long long i = (long long)blockIdx.x * blockDim.x + threadIdx.x; i < n128; i += stride) {
        uint4 v = in[i];
        const unsigned int uu[4] = {v.x, v.y, v.z, v.w};
        float f[8];
#pragma unroll
        for (int j = 0; j < 4; ++j) {
            f[2 * j]     = __uint_as_float((uu[j] & 0xFFFFu) << 16);
            f[2 * j + 1] = __uint_as_float(uu[j] & 0xFFFF0000u);
        }
#pragma unroll
        for (int j = 0; j < 8; ++j)
            f[j] = fminf(fmaxf(f[j] * s, -FP8MAXF), FP8MAXF);
#if defined(HAVE_CVT_PK_FP8)
        int r0 = __builtin_amdgcn_cvt_pk_fp8_f32(f[0], f[1], 0, false);
        r0     = __builtin_amdgcn_cvt_pk_fp8_f32(f[2], f[3], r0, true);
        int r1 = __builtin_amdgcn_cvt_pk_fp8_f32(f[4], f[5], 0, false);
        r1     = __builtin_amdgcn_cvt_pk_fp8_f32(f[6], f[7], r1, true);
        outq[i] = make_uint2((unsigned int)r0, (unsigned int)r1);
#else
        outq[i] = make_uint2(sw_pack4_e4m3(f[0], f[1], f[2], f[3]),
                             sw_pack4_e4m3(f[4], f[5], f[6], f[7]));
#endif
    }
}

__global__ __launch_bounds__(256, 1) void fp8_gemm_wmma_kernel(
    const unsigned char* __restrict__ xq, const unsigned char* __restrict__ wq,
    const float* __restrict__ sf, __hip_bfloat16* __restrict__ out) {
    const int lane = threadIdx.x & 31;
    const int wave = threadIdx.x >> 5;
    const int hl   = lane >> 4;
    const int l16  = lane & 15;

    const int tileN = blockIdx.x * 128 + (wave & 3) * 32;
    const int tileM = blockIdx.y * 128 + (wave >> 2) * 64;

    v8f acc[4][2] = {};

    for (int k0 = 0; k0 < KDIM; k0 += 128) {
        if (k0 + 128 < KDIM) {
            __builtin_prefetch(xq + (size_t)(tileM + l16) * KDIM + k0 + 128, 0, 3);
            __builtin_prefetch(wq + (size_t)(tileN + l16) * KDIM + k0 + 128, 0, 3);
        }
        v16i a[4];
        v16i b[2];
#pragma unroll
        for (int mi = 0; mi < 4; ++mi) {
            const unsigned long long* p = (const unsigned long long*)
                (xq + (size_t)(tileM + 16 * mi + l16) * KDIM + k0 + hl * 16);
#pragma unroll
            for (int i = 0; i < 8; ++i) {
                unsigned long long d = p[4 * (i >> 1) + (i & 1)];
                a[mi][2 * i]     = (int)(unsigned int)d;
                a[mi][2 * i + 1] = (int)(unsigned int)(d >> 32);
            }
        }
#pragma unroll
        for (int ni = 0; ni < 2; ++ni) {
            const uint4* q = (const uint4*)
                (wq + (size_t)(tileN + 16 * ni + l16) * KDIM + k0 + hl * 16);
#pragma unroll
            for (int j = 0; j < 4; ++j) {
                uint4 d = q[2 * j];
                b[ni][4 * j]     = (int)d.x;
                b[ni][4 * j + 1] = (int)d.y;
                b[ni][4 * j + 2] = (int)d.z;
                b[ni][4 * j + 3] = (int)d.w;
            }
        }
#pragma unroll
        for (int mi = 0; mi < 4; ++mi)
#pragma unroll
            for (int ni = 0; ni < 2; ++ni)
                acc[mi][ni] = __builtin_amdgcn_wmma_f32_16x16x128_fp8_fp8(
                    a[mi], b[ni], (short)0, acc[mi][ni], false, false);
    }

    const float s = sf[4];
#pragma unroll
    for (int mi = 0; mi < 4; ++mi) {
#pragma unroll
        for (int ni = 0; ni < 2; ++ni) {
            const int col = tileN + 16 * ni + l16;
#pragma unroll
            for (int r = 0; r < 8; ++r) {
                const int row = tileM + 16 * mi + hl * 8 + r;
                out[(size_t)row * NDIM + col] = __float2bfloat16(acc[mi][ni][r] * s);
            }
        }
    }
}

extern "C" void kernel_launch(void* const* d_in, const int* in_sizes, int n_in,
                              void* d_out, int out_size, void* d_ws, size_t ws_size,
                              hipStream_t stream) {
    (void)in_sizes; (void)n_in; (void)out_size; (void)ws_size;
    const uint4* x = (const uint4*)d_in[0];
    const uint4* w = (const uint4*)d_in[1];
    __hip_bfloat16* out = (__hip_bfloat16*)d_out;

    unsigned char* ws   = (unsigned char*)d_ws;
    unsigned int*  scal = (unsigned int*)ws;
    unsigned char* xq   = ws + 256;
    unsigned char* wq   = ws + 256 + (size_t)MDIM * KDIM;

    const long long nx128 = (long long)MDIM * KDIM / 8;
    const long long nw128 = (long long)NDIM * KDIM / 8;

    hipLaunchKernelGGL(init_scalars_kernel, dim3(1), dim3(32), 0, stream, scal);
    hipLaunchKernelGGL(amax_bf16_kernel, dim3(2048), dim3(256), 0, stream, x, nx128, scal + 0);
    hipLaunchKernelGGL(amax_bf16_kernel, dim3(2048), dim3(256), 0, stream, w, nw128, scal + 1);
    hipLaunchKernelGGL(compute_scales_kernel, dim3(1), dim3(1), 0, stream, scal);
    hipLaunchKernelGGL(quantize_bf16_fp8_kernel, dim3(4096), dim3(256), 0, stream,
                       x, (uint2*)xq, nx128, (const float*)scal, 2);
    hipLaunchKernelGGL(quantize_bf16_fp8_kernel, dim3(4096), dim3(256), 0, stream,
                       w, (uint2*)wq, nw128, (const float*)scal, 3);
    hipLaunchKernelGGL(fp8_gemm_wmma_kernel, dim3(NDIM / 128, MDIM / 128), dim3(256), 0, stream,
                       xq, wq, (const float*)scal, out);
}
